// LSSModel_29652454211772
// MI455X (gfx1250) — matrix operand order repaired and checked by reading, not compiled, not run on hardware
//
#include <hip/hip_runtime.h>
#include <hip/hip_bf16.h>

// ---------------------------------------------------------------------------
// LSS BEV pipeline for gfx1250 (MI455X). All convs are implicit-GEMM with
// v_wmma_f32_16x16x32_f16; weight tiles are DMA'd to LDS by the Tensor Data
// Mover (tensor_load_to_lds + s_wait_tensorcnt) when the builtin exists.
// ---------------------------------------------------------------------------

typedef _Float16 v16h __attribute__((ext_vector_type(16)));
typedef _Float16 v8h  __attribute__((ext_vector_type(8)));
typedef float    v8f  __attribute__((ext_vector_type(8)));
typedef float    f4   __attribute__((ext_vector_type(4)));
typedef unsigned int u32x4 __attribute__((ext_vector_type(4)));
typedef int          i32x4 __attribute__((ext_vector_type(4)));
typedef int          i32x8 __attribute__((ext_vector_type(8)));

#if defined(__has_builtin)
# if __has_builtin(__builtin_amdgcn_tensor_load_to_lds)
#  define HAVE_TDM 1
# endif
#endif

#define TM 128
#define TN 64
#define TK 32
#define AS_STRIDE 40    // f16 A tile: 80B row pitch -> conflict-free
#define BSF_STRIDE 36   // f32 B tile: 144B pitch (16B aligned, banks distinct)
#define CS_STRIDE 66    // f32 epilogue staging

// ---------------------------------------------------------------------------
// Generic conv (NCHW, OIHW weights) as implicit GEMM:
//   M = N*Ho*Wo, Ncol = Co, K = Ci*KH*KW  (weight tile is wgt[co*Kt + k])
// Workgroup: 256 threads = 8 waves; tile 128x64; K stepped by 32.
// ---------------------------------------------------------------------------
template <int KH, int KW>
__global__ __launch_bounds__(256)
void conv_wmma(const float* __restrict__ in, const float* __restrict__ wgt,
               const float* __restrict__ bias, float* __restrict__ out,
               int Nimg, int Ci, int Hi, int Wi, int Co, int Ho, int Wo,
               int stride, int padh, int padw, int do_relu)
{
    __shared__ _Float16 As[TM * AS_STRIDE];
    __shared__ float    Bsf[TN * BSF_STRIDE];  // [co][k], f32 (TDM target)
    __shared__ float    Cs[TM * CS_STRIDE];    // epilogue staging

    const int tid  = threadIdx.x;
    const int wave = tid >> 5;
    const int lane = tid & 31;
    const int m0 = blockIdx.x * TM;
    const int n0 = blockIdx.y * TN;
    const int M   = Nimg * Ho * Wo;
    const int HWo = Ho * Wo;
    constexpr int KHW = KH * KW;
    const int Kt = Ci * KHW;

    // ---- per-thread A-row decomposition (constant across the K loop) ----
    const int arowFill = tid >> 1;               // A row this thread fills
    const int akOff    = (tid & 1) * 16;         // first of 16 consecutive k
    int a_ih0 = 0, a_iw0 = 0;
    const float* a_base = nullptr;
    bool a_mvalid = false;
    {
        int m = m0 + arowFill;
        if (m < M) {
            int n  = m / HWo; int rm = m - n * HWo;
            int ho = rm / Wo; int wo = rm - ho * Wo;
            a_ih0 = ho * stride - padh;
            a_iw0 = wo * stride - padw;
            a_base = in + ((long)n * Ci) * Hi * Wi;
            a_mvalid = true;
        }
    }
#ifndef HAVE_TDM
    const int browFill = tid >> 2;               // B row (co offset) filled
    const int bkOff    = (tid & 3) * 8;
    const bool b_nvalid = (n0 + browFill) < Co;
    const float* b_base = wgt + (long)(n0 + browFill) * Kt;
#endif

    v8f acc[4];
    #pragma unroll
    for (int i = 0; i < 4; ++i)
        #pragma unroll
        for (int j = 0; j < 8; ++j) acc[i][j] = 0.0f;

    for (int k0 = 0; k0 < Kt; k0 += TK) {
        // ---- stage A: 128x32 patch tile (im2col on the fly, f32 -> f16) ----
        #pragma unroll
        for (int i = 0; i < 16; ++i) {
            int k = k0 + akOff + i;
            float v = 0.0f;
            if (a_mvalid && k < Kt) {
                int ci = k / KHW; int rk = k - ci * KHW;   // constant divisor
                int kh = rk / KW; int kw = rk - kh * KW;
                int ih = a_ih0 + kh;
                int iw = a_iw0 + kw;
                if (ih >= 0 && ih < Hi && iw >= 0 && iw < Wi)
                    v = a_base[((long)ci * Hi + ih) * Wi + iw];
            }
            As[arowFill * AS_STRIDE + akOff + i] = (_Float16)v;
        }

        // ---- stage B: 64x32 f32 weight tile ----
#ifdef HAVE_TDM
        // Tensor Data Mover: one wave DMAs the tile into LDS with row padding
        // (pad 4 dwords every 32 -> LDS row stride 36 dwords).
        if (wave == 0) {
            unsigned long long gaddr =
                (unsigned long long)(unsigned long long)(wgt + (size_t)n0 * (size_t)Kt + (size_t)k0);
            unsigned ldsoff = (unsigned)(unsigned long long)&Bsf[0];
            int rem0 = Kt - k0;          // K remaining (OOB -> zero fill)
            int rem1 = Co - n0;          // Co remaining
            u32x4 g0;
            g0[0] = 1u;                                              // count=1
            g0[1] = ldsoff;                                          // lds_addr
            g0[2] = (unsigned)gaddr;                                 // gaddr lo
            g0[3] = (unsigned)((gaddr >> 32) & 0x1FFFFFFull) | (2u << 30); // type=2
            i32x8 g1;
            g1[0] = (2 << 16) | (1 << 20) | (4 << 22) | (3 << 25);   // 4B, pad 4dw/32dw
            g1[1] = (rem0 & 0xFFFF) << 16;                           // tensor_dim0 lo
            g1[2] = ((rem0 >> 16) & 0xFFFF) | ((rem1 & 0xFFFF) << 16);
            g1[3] = ((rem1 >> 16) & 0xFFFF) | (TK << 16);            // tile_dim0=32
            g1[4] = TN;                                              // tile_dim1=64
            g1[5] = Kt;                                              // dim0 stride lo
            g1[6] = 0;
            g1[7] = 0;
            i32x4 z4 = {0, 0, 0, 0};
#if __clang_major__ >= 23
            i32x8 z8 = {0, 0, 0, 0, 0, 0, 0, 0};
            __builtin_amdgcn_tensor_load_to_lds(g0, g1, z4, z4, z8, 0);
#else
            __builtin_amdgcn_tensor_load_to_lds(g0, g1, z4, z4, 0);
#endif
#if __has_builtin(__builtin_amdgcn_s_wait_tensorcnt)
            __builtin_amdgcn_s_wait_tensorcnt(0);
#else
            asm volatile("s_wait_tensorcnt 0x0" ::: "memory");
#endif
        }
#else
        #pragma unroll
        for (int i = 0; i < 8; ++i) {
            int k = k0 + bkOff + i;
            float v = (b_nvalid && k < Kt) ? b_base[k] : 0.0f;
            Bsf[browFill * BSF_STRIDE + bkOff + i] = v;
        }
        if (b_nvalid && (k0 + TK) < Kt)
            __builtin_prefetch(&b_base[k0 + TK + bkOff], 0, 3);
#endif
        __syncthreads();

        // ---- A fragment: ISA 16-bit A layout (lanes<16: K 0..7,16..23) ----
        int arow  = wave * 16 + (lane & 15);
        int abase = (lane >= 16) ? 8 : 0;
        v8h alo = *(v8h*)&As[arow * AS_STRIDE + abase];
        v8h ahi = *(v8h*)&As[arow * AS_STRIDE + abase + 16];
        v16h a;
        #pragma unroll
        for (int j = 0; j < 8; ++j) { a[j] = alo[j]; a[j + 8] = ahi[j]; }

        // ---- 4 WMMAs reusing A across the 64-wide N tile ----
        int bbase = (lane >= 16) ? 8 : 0;
        #pragma unroll
        for (int nt = 0; nt < 4; ++nt) {
            int bcol = nt * 16 + (lane & 15);
            const float* bp = &Bsf[bcol * BSF_STRIDE + bbase];
            f4 b0 = *(const f4*)(bp);
            f4 b1 = *(const f4*)(bp + 4);
            f4 b2 = *(const f4*)(bp + 16);
            f4 b3 = *(const f4*)(bp + 20);
            v16h b;
            #pragma unroll
            for (int j = 0; j < 4; ++j) {
                b[j]      = (_Float16)b0[j];
                b[j + 4]  = (_Float16)b1[j];
                b[j + 8]  = (_Float16)b2[j];
                b[j + 12] = (_Float16)b3[j];
            }
            acc[nt] = __builtin_amdgcn_wmma_f32_16x16x32_f16(
                false, a, false, b, (short)0, acc[nt], false, false);
        }
        __syncthreads();
    }

    // ---- stage C in LDS (C/D layout: VGPR r -> M=r / r+8; lane -> N) ----
    #pragma unroll
    for (int nt = 0; nt < 4; ++nt) {
        int col = nt * 16 + (lane & 15);
        #pragma unroll
        for (int r = 0; r < 8; ++r) {
            int mloc = wave * 16 + r + ((lane >= 16) ? 8 : 0);
            Cs[mloc * CS_STRIDE + col] = acc[nt][r];
        }
    }
    __syncthreads();

    // ---- coalesced epilogue: lanes walk M (contiguous rm) ----
    const int row = tid & 127;                   // constant per thread
    int em = m0 + row;
    if (em < M) {
        int n = em / HWo; int rm = em - n * HWo;
        float* orow = out + (long)n * Co * HWo + rm;
        #pragma unroll
        for (int it = 0; it < 32; ++it) {
            int col = it * 2 + (tid >> 7);
            int co = n0 + col;
            if (co < Co) {
                float v = Cs[row * CS_STRIDE + col];
                if (bias) v += bias[co];
                if (do_relu) v = fmaxf(v, 0.0f);
                orow[(long)co * HWo] = v;
            }
        }
    }
}

// ---------------------------------------------------------------------------
// Per-channel batch-norm statistics (mean over N,H,W) -> mean[c], istd[c]
// ---------------------------------------------------------------------------
__global__ __launch_bounds__(256)
void channel_stats(const float* __restrict__ x, float* __restrict__ mean,
                   float* __restrict__ istd, int Nimg, int C, int HW)
{
    int c = blockIdx.x;
    float s = 0.0f, sq = 0.0f;
    for (int n = 0; n < Nimg; ++n) {
        const float* p = x + ((long)n * C + c) * HW;
        for (int r = threadIdx.x; r < HW; r += 256) {
            float v = p[r];
            s += v; sq += v * v;
        }
    }
    __shared__ float ss[256], ssq[256];
    ss[threadIdx.x] = s; ssq[threadIdx.x] = sq;
    __syncthreads();
    for (int o = 128; o > 0; o >>= 1) {
        if ((int)threadIdx.x < o) {
            ss[threadIdx.x]  += ss[threadIdx.x + o];
            ssq[threadIdx.x] += ssq[threadIdx.x + o];
        }
        __syncthreads();
    }
    if (threadIdx.x == 0) {
        float cnt = (float)Nimg * (float)HW;
        float m = ss[0] / cnt;
        float v = ssq[0] / cnt - m * m;
        mean[c] = m;
        istd[c] = rsqrtf(v + 1e-5f);
    }
}

// grid: (ceil(HW/256), Nimg*C); channel = blockIdx.y % C
__global__ void bn_relu_k(float* __restrict__ x, const float* __restrict__ mean,
                          const float* __restrict__ istd, const float* __restrict__ g,
                          const float* __restrict__ b, int C, int HW)
{
    int plane = blockIdx.y;
    int c = plane % C;
    int r = blockIdx.x * blockDim.x + threadIdx.x;
    if (r >= HW) return;
    float* p = x + (long)plane * HW;
    float v = (p[r] - mean[c]) * istd[c] * g[c] + b[c];
    p[r] = fmaxf(v, 0.0f);
}

// ---------------------------------------------------------------------------
// Softmax over channel (depth) dimension: x is (NI, Dd, HW), in place.
// ---------------------------------------------------------------------------
__global__ void softmax_depth(float* __restrict__ x, int NI, int Dd, int HW)
{
    int i = blockIdx.x * blockDim.x + threadIdx.x;
    if (i >= NI * HW) return;
    int n = i / HW, r = i - n * HW;
    float* p = x + (long)n * Dd * HW + r;
    float mx = -1e30f;
    for (int d = 0; d < Dd; ++d) mx = fmaxf(mx, p[(long)d * HW]);
    float s = 0.0f;
    for (int d = 0; d < Dd; ++d) { float e = __expf(p[(long)d * HW] - mx); p[(long)d * HW] = e; s += e; }
    float inv = 1.0f / s;
    for (int d = 0; d < Dd; ++d) p[(long)d * HW] *= inv;
}

// ---------------------------------------------------------------------------
// Geometry: combine = rot @ inv(K), trans. One thread per (b, cam).
// ---------------------------------------------------------------------------
#define BB 4
#define NCAM 6
#define DD 41
#define FHH 14
#define FWW 25
#define NXX 200
#define NYY 200
#define CBEV 64

__global__ void geom_setup(const float* __restrict__ intr, const float* __restrict__ extr,
                           float* __restrict__ comb)
{
    int i = blockIdx.x * blockDim.x + threadIdx.x;
    if (i >= BB * NCAM) return;
    const float* K = intr + i * 9;
    const float* E = extr + i * 16;
    float a = K[0], b = K[1], c = K[2], d = K[3], e = K[4], f = K[5],
          g = K[6], h = K[7], k = K[8];
    float A =  e * k - f * h;
    float Bm = -(d * k - f * g);
    float Cm =  d * h - e * g;
    float det = a * A + b * Bm + c * Cm;
    float id = 1.0f / det;
    float inv[9] = { A * id,            -(b * k - c * h) * id,  (b * f - c * e) * id,
                     Bm * id,            (a * k - c * g) * id, -(a * f - c * d) * id,
                     Cm * id,           -(a * h - b * g) * id,  (a * e - b * d) * id };
    float* o = comb + i * 12;
    for (int r = 0; r < 3; ++r) {
        for (int cc = 0; cc < 3; ++cc) {
            float s = 0.0f;
            for (int t = 0; t < 3; ++t) s += E[r * 4 + t] * inv[t * 3 + cc];
            o[r * 3 + cc] = s;
        }
        o[9 + r] = E[r * 4 + 3];
    }
}

// ---------------------------------------------------------------------------
// Splat: one wave per frustum point; 64 channels / 32 lanes = 2 atomics each.
// bev is NCHW (B, C_BEV, NY, NX).
// ---------------------------------------------------------------------------
__global__ __launch_bounds__(256)
void splat(const float* __restrict__ depth, const float* __restrict__ fc,
           const float* __restrict__ comb, float* __restrict__ bev)
{
    const long P = (long)BB * NCAM * DD * FHH * FWW;
    int wave = threadIdx.x >> 5, lane = threadIdx.x & 31;
    long p = (long)blockIdx.x * 8 + wave;
    if (p >= P) return;
    int t = (int)p;
    int wv  = t % FWW; t /= FWW;
    int hh  = t % FHH; t /= FHH;
    int dd  = t % DD;  t /= DD;
    int cam = t % NCAM; t /= NCAM;
    int b = t;

    float dpt = 4.0f + (float)dd;
    float xs = (float)wv * (399.0f / 24.0f);
    float ys = (float)hh * (223.0f / 13.0f);
    float px = xs * dpt, py = ys * dpt, pz = dpt;
    const float* cm = comb + (b * NCAM + cam) * 12;
    float gx = cm[0] * px + cm[1] * py + cm[2] * pz + cm[9];
    float gy = cm[3] * px + cm[4] * py + cm[5] * pz + cm[10];
    float gz = cm[6] * px + cm[7] * py + cm[8] * pz + cm[11];
    int ix = (int)floorf((gx + 50.0f) * 2.0f);
    int iy = (int)floorf((gy + 50.0f) * 2.0f);
    int iz = (int)floorf((gz + 10.0f) * (1.0f / 20.0f));
    if (ix < 0 || ix >= NXX || iy < 0 || iy >= NYY || iz < 0 || iz >= 1) return;

    int img = b * NCAM + cam;
    float dval = depth[(((long)img * DD + dd) * FHH + hh) * FWW + wv];
    long fbase = ((long)img * CBEV) * (FHH * FWW) + (long)hh * FWW + wv;
    long bbase = ((long)b * CBEV) * (NYY * NXX) + (long)iy * NXX + ix;
    for (int c = lane; c < CBEV; c += 32) {
        float v = dval * fc[fbase + (long)c * (FHH * FWW)];
        atomicAdd(&bev[bbase + (long)c * (NYY * NXX)], v);
    }
}

__global__ void zero_k(float* __restrict__ p, long n)
{
    long i = (long)blockIdx.x * blockDim.x + threadIdx.x;
    if (i < n) p[i] = 0.0f;
}

// ---------------------------------------------------------------------------
// Host orchestration
// ---------------------------------------------------------------------------
static void launch_conv(const float* in, const float* w, const float* bias, float* out,
                        int N, int Ci, int Hi, int Wi, int Co, int Ho, int Wo,
                        int KH, int s, int ph, int pw, int relu, hipStream_t stream)
{
    int M = N * Ho * Wo;
    dim3 g((M + TM - 1) / TM, (Co + TN - 1) / TN);
    if (KH == 7)
        conv_wmma<7, 7><<<g, 256, 0, stream>>>(in, w, bias, out, N, Ci, Hi, Wi,
                                               Co, Ho, Wo, s, ph, pw, relu);
    else if (KH == 3)
        conv_wmma<3, 3><<<g, 256, 0, stream>>>(in, w, bias, out, N, Ci, Hi, Wi,
                                               Co, Ho, Wo, s, ph, pw, relu);
    else
        conv_wmma<1, 1><<<g, 256, 0, stream>>>(in, w, bias, out, N, Ci, Hi, Wi,
                                               Co, Ho, Wo, s, ph, pw, relu);
}

static void launch_bn_relu(float* x, const float* g, const float* b, float* mean, float* istd,
                           int N, int C, int HW, hipStream_t stream)
{
    channel_stats<<<C, 256, 0, stream>>>(x, mean, istd, N, C, HW);
    dim3 grid((HW + 255) / 256, N * C);
    bn_relu_k<<<grid, 256, 0, stream>>>(x, mean, istd, g, b, C, HW);
}

extern "C" void kernel_launch(void* const* d_in, const int* in_sizes, int n_in,
                              void* d_out, int out_size, void* d_ws, size_t ws_size,
                              hipStream_t stream)
{
    (void)in_sizes; (void)n_in; (void)out_size; (void)ws_size;

    const float* images     = (const float*)d_in[0];
    const float* intrinsics = (const float*)d_in[1];
    const float* extrinsics = (const float*)d_in[2];
    const float* w_stem = (const float*)d_in[3];
    const float* g_stem = (const float*)d_in[4];
    const float* b_stem = (const float*)d_in[5];
    const float* w1 = (const float*)d_in[6];
    const float* g1 = (const float*)d_in[7];
    const float* b1 = (const float*)d_in[8];
    const float* w2 = (const float*)d_in[9];
    const float* g2 = (const float*)d_in[10];
    const float* b2 = (const float*)d_in[11];
    const float* w3 = (const float*)d_in[12];
    const float* g3 = (const float*)d_in[13];
    const float* b3 = (const float*)d_in[14];
    const float* w_neck = (const float*)d_in[15];
    const float* g_neck = (const float*)d_in[16];
    const float* b_neck = (const float*)d_in[17];
    const float* w_depth    = (const float*)d_in[18];
    const float* bias_depth = (const float*)d_in[19];
    const float* w_comp     = (const float*)d_in[20];
    const float* bias_comp  = (const float*)d_in[21];
    const float* w_bev1 = (const float*)d_in[22];
    const float* g_bev1 = (const float*)d_in[23];
    const float* b_bev1 = (const float*)d_in[24];
    const float* w_bev2 = (const float*)d_in[25];
    const float* g_bev2 = (const float*)d_in[26];
    const float* b_bev2 = (const float*)d_in[27];

    float* ws = (float*)d_ws;
    size_t off = 0;
    auto take = [&](size_t n) { float* p = ws + off; off += n; return p; };

    const int NI = BB * NCAM;                       // 24 images
    float* buf0 = take((size_t)NI * 64 * 112 * 200);    // stem out
    float* buf1 = take((size_t)NI * 256 * 56 * 100);    // layer1 out
    float* buf2 = take((size_t)NI * 512 * 28 * 50);     // layer2 out
    float* buf3 = take((size_t)NI * 1024 * 14 * 25);    // layer3 out
    float* bufn = take((size_t)NI * 512 * 14 * 25);     // neck out
    float* bufd = take((size_t)NI * DD * 14 * 25);      // depth logits/softmax
    float* bufc = take((size_t)NI * CBEV * 14 * 25);    // comp features
    float* comb = take((size_t)NI * 12);                // combine + trans
    float* bev0 = take((size_t)BB * CBEV * NYY * NXX);  // splat accumulator
    float* bev1 = take((size_t)BB * CBEV * NYY * NXX);  // bev conv1 out
    float* meanb = take(1024);
    float* istdb = take(1024);
    float* outp = (float*)d_out;

    // --- backbone (each: WMMA conv -> batch stats -> in-place BN+ReLU) ---
    launch_conv(images, w_stem, nullptr, buf0, NI, 3, 224, 400, 64, 112, 200, 7, 2, 2, 2, 0, stream);
    launch_bn_relu(buf0, g_stem, b_stem, meanb, istdb, NI, 64, 112 * 200, stream);

    launch_conv(buf0, w1, nullptr, buf1, NI, 64, 112, 200, 256, 56, 100, 3, 2, 0, 0, 0, stream);
    launch_bn_relu(buf1, g1, b1, meanb, istdb, NI, 256, 56 * 100, stream);

    launch_conv(buf1, w2, nullptr, buf2, NI, 256, 56, 100, 512, 28, 50, 3, 2, 0, 0, 0, stream);
    launch_bn_relu(buf2, g2, b2, meanb, istdb, NI, 512, 28 * 50, stream);

    launch_conv(buf2, w3, nullptr, buf3, NI, 512, 28, 50, 1024, 14, 25, 3, 2, 0, 0, 0, stream);
    launch_bn_relu(buf3, g3, b3, meanb, istdb, NI, 1024, 14 * 25, stream);

    launch_conv(buf3, w_neck, nullptr, bufn, NI, 1024, 14, 25, 512, 14, 25, 3, 1, 1, 1, 0, stream);
    launch_bn_relu(bufn, g_neck, b_neck, meanb, istdb, NI, 512, 14 * 25, stream);

    // --- heads ---
    launch_conv(bufn, w_depth, bias_depth, bufd, NI, 512, 14, 25, DD, 14, 25, 1, 1, 0, 0, 0, stream);
    {
        int pix = NI * 14 * 25;
        softmax_depth<<<(pix + 255) / 256, 256, 0, stream>>>(bufd, NI, DD, 14 * 25);
    }
    launch_conv(bufn, w_comp, bias_comp, bufc, NI, 512, 14, 25, CBEV, 14, 25, 1, 1, 0, 0, 1, stream);

    // --- geometry + splat ---
    geom_setup<<<1, 32, 0, stream>>>(intrinsics, extrinsics, comb);
    {
        long nbev = (long)BB * CBEV * NYY * NXX;
        zero_k<<<(int)((nbev + 255) / 256), 256, 0, stream>>>(bev0, nbev);
        long P = (long)BB * NCAM * DD * FHH * FWW;       // 344,400 points
        splat<<<(int)((P + 7) / 8), 256, 0, stream>>>(bufd, bufc, comb, bev0);
    }

    // --- BEV head ---
    launch_conv(bev0, w_bev1, nullptr, bev1, BB, CBEV, NYY, NXX, CBEV, NYY, NXX, 3, 1, 1, 1, 0, stream);
    launch_bn_relu(bev1, g_bev1, b_bev1, meanb, istdb, BB, CBEV, NYY * NXX, stream);

    launch_conv(bev1, w_bev2, nullptr, outp, BB, CBEV, NYY, NXX, CBEV, NYY, NXX, 3, 1, 1, 1, 0, stream);
    launch_bn_relu(outp, g_bev2, b_bev2, meanb, istdb, BB, CBEV, NYY * NXX, stream);
}
